// Net_23021024706961
// MI455X (gfx1250) — matrix operand order repaired and checked by reading, not compiled, not run on hardware
//
#include <hip/hip_runtime.h>

typedef __attribute__((ext_vector_type(16))) _Float16 v16h;
typedef __attribute__((ext_vector_type(8)))  float    v8f;

constexpr int  cB  = 8;
constexpr int  cN  = 1024;
constexpr int  cK  = 20;
constexpr long cNK = (long)cN * cK;   // 20480 (multiple of 16)

__device__ __forceinline__ float lrelu02(float v) { return v < 0.f ? 0.2f * v : v; }

// ====================================================================
// Generic strided WMMA GEMM:  per batch b:  Y[o,p] = sum_c A[o,c]*X[c,p]
// One wave (32 lanes) computes a 16x16 D tile via v_wmma_f32_16x16x32_f16.
// A/X tiles are cooperatively staged into LDS (coalesced, unguarded on
// the fast path); fragments are built from LDS per ISA 7.12.2 layouts.
// ====================================================================
__global__ __launch_bounds__(32)
void wmma_gemm_strided(const float* __restrict__ A, const float* __restrict__ X,
                       float* __restrict__ Y, int O, int C, int P,
                       long sAB, long sAO, long sAC, long sXB, long sXC,
                       long sYB, long sYO)
{
    (void)P;
    __shared__ float lA[16*32];   // A tile: 16 rows (o) x 32 cols (k)
    __shared__ float lX[32*16];   // X tile: 32 rows (k) x 16 cols (p)
    const int lane = threadIdx.x;
    const int row  = lane & 15;
    const int grp  = lane >> 4;
    const int p0   = blockIdx.x * 16;
    const int o0   = blockIdx.y * 16;
    const int b    = blockIdx.z;
    const float* Ab = A + (long)b * sAB;
    const float* Xb = X + (long)b * sXB;
    v8f acc = {};
    for (int k0 = 0; k0 < C; k0 += 32) {
        const bool fast = (o0 + 16 <= O) && (k0 + 32 <= C);
        if (fast) {
            for (int t = lane; t < 512; t += 32) {
                const int rA = t >> 5, cA = t & 31;
                lA[t] = Ab[(long)(o0 + rA)*sAO + (long)(k0 + cA)*sAC];
                const int rX = t >> 4, cX = t & 15;
                lX[t] = Xb[(long)(k0 + rX)*sXC + p0 + cX];
            }
        } else {
            for (int t = lane; t < 512; t += 32) {
                const int rA = t >> 5, cA = t & 31;
                const int kA = k0 + cA;
                lA[t] = ((o0 + rA) < O && kA < C)
                        ? Ab[(long)(o0 + rA)*sAO + (long)kA*sAC] : 0.f;
                const int rX = t >> 4, cX = t & 15;
                const int kX = k0 + rX;
                lX[t] = (kX < C) ? Xb[(long)kX*sXC + p0 + cX] : 0.f;
            }
        }
        __syncthreads();
        v16h af, bf;
        #pragma unroll
        for (int e = 0; e < 16; ++e) {
            const int v  = e >> 1, h = e & 1;
            const int ka = (v < 4 ? 2*v + h : 16 + 2*(v-4) + h) + (grp ? 8 : 0);
            af[e] = (_Float16)lA[row*32 + ka];
            const int kb = (e < 8 ? e : e + 8) + (grp ? 8 : 0);
            bf[e] = (_Float16)lX[kb*16 + row];
        }
        acc = __builtin_amdgcn_wmma_f32_16x16x32_f16(false, af, false, bf,
                                                     (short)0, acc, false, false);
        __syncthreads();
    }
    float* Yb = Y + (long)b * sYB;
    #pragma unroll
    for (int r = 0; r < 8; ++r) {
        const int m = o0 + r + 8*grp;
        if (m < O) Yb[(long)m*sYO + p0 + row] = acc[r];
    }
}

// ====================================================================
// Fused attention->xo:  aw = Wa(384,fc) @ attn_tile(fc,16)  (WMMA, LDS),
// then xo[p,o] = sum_i aw[o*6+i,p] * (W_efr_p[i]*fmp[p])
// attn tile in LDS is zero-padded to a multiple of 32 rows so B-fragment
// reads need no guards.
// ====================================================================
__global__ __launch_bounds__(32)
void gct_xo_wmma(const float* __restrict__ Wa, const float* __restrict__ attn,
                 const float* __restrict__ Wep, const float* __restrict__ fmp,
                 float* __restrict__ xo, int fc)
{
    __shared__ float sA[128*16];
    __shared__ float aw[384*16];
    __shared__ float vv[6*16];
    const int lane = threadIdx.x;
    const int row  = lane & 15, grp = lane >> 4;
    const int b    = blockIdx.y;
    const int p0   = blockIdx.x * 16;
    const int fcp  = (fc + 31) & ~31;
    for (int t = lane; t < fcp*16; t += 32) {
        int j = t >> 4, pl = t & 15;
        sA[t] = (j < fc) ? attn[((long)b*fc + j)*cNK + p0 + pl] : 0.f;
    }
    for (int t = lane; t < 96; t += 32) {
        int i = t >> 4, pl = t & 15;
        vv[t] = Wep[i] * fmp[(long)b*cNK + p0 + pl];
    }
    __syncthreads();
    for (int ot = 0; ot < 24; ++ot) {
        v8f acc = {};
        const int oA = ot*16 + row;
        for (int k0 = 0; k0 < fc; k0 += 32) {
            v16h af, bf;
            #pragma unroll
            for (int e = 0; e < 16; ++e) {
                const int v = e >> 1, h = e & 1;
                const int kA = k0 + (v < 4 ? 2*v + h : 16 + 2*(v-4) + h) + (grp ? 8 : 0);
                af[e] = (_Float16)((kA < fc) ? Wa[(long)oA*fc + kA] : 0.f);
                const int kB = k0 + (e < 8 ? e : e + 8) + (grp ? 8 : 0);
                bf[e] = (_Float16)sA[kB*16 + row];
            }
            acc = __builtin_amdgcn_wmma_f32_16x16x32_f16(false, af, false, bf,
                                                         (short)0, acc, false, false);
        }
        #pragma unroll
        for (int r = 0; r < 8; ++r) aw[(ot*16 + r + 8*grp)*16 + row] = acc[r];
    }
    __syncthreads();
    for (int t = lane; t < 1024; t += 32) {
        int pl = t >> 6, o = t & 63;
        float s = 0.f;
        #pragma unroll
        for (int i = 0; i < 6; ++i) s += aw[(o*6 + i)*16 + pl] * vv[i*16 + pl];
        xo[((long)b*64 + o)*cNK + p0 + pl] = s;
    }
}

// ====================================================================
// Fused edge-feature (ggf) conv:  gather X tile into LDS, WMMA GEMM.
// ggf channel c<Cf : F[b,c,idx]-F[b,c,n] ; c>=Cf : F[b,c-Cf,n]
// (Cg = 2*Cf is always a multiple of 32; Cout a multiple of 16.)
// ====================================================================
__global__ __launch_bounds__(32)
void ggf_conv_wmma(const float* __restrict__ Wc, const float* __restrict__ F,
                   const int* __restrict__ idx, float* __restrict__ out,
                   int Cf, int Cout)
{
    __shared__ float xt[256*16];
    const int lane = threadIdx.x;
    const int row  = lane & 15, grp = lane >> 4;
    const int b    = blockIdx.y;
    const int p0   = blockIdx.x * 16;
    const int Cg   = 2*Cf;
    for (int t = lane; t < Cg*16; t += 32) {
        int c = t >> 4, pl = t & 15;
        long p = p0 + pl;
        int n  = (int)(p / cK);
        int j  = idx[(long)b*cNK + p];
        float v;
        if (c < Cf) v = F[((long)b*Cf + c)*cN + j] - F[((long)b*Cf + c)*cN + n];
        else        v = F[((long)b*Cf + (c-Cf))*cN + n];
        xt[c*16 + pl] = v;
    }
    __syncthreads();
    for (int ot = 0; ot < Cout/16; ++ot) {
        v8f acc = {};
        const int oA = ot*16 + row;
        for (int k0 = 0; k0 < Cg; k0 += 32) {
            v16h af, bf;
            #pragma unroll
            for (int e = 0; e < 16; ++e) {
                const int v = e >> 1, h = e & 1;
                const int kA = k0 + (v < 4 ? 2*v + h : 16 + 2*(v-4) + h) + (grp ? 8 : 0);
                af[e] = (_Float16)(Wc[(long)oA*Cg + kA]);
                const int kB = k0 + (e < 8 ? e : e + 8) + (grp ? 8 : 0);
                bf[e] = (_Float16)(xt[kB*16 + row]);
            }
            acc = __builtin_amdgcn_wmma_f32_16x16x32_f16(false, af, false, bf,
                                                         (short)0, acc, false, false);
        }
        #pragma unroll
        for (int r = 0; r < 8; ++r) {
            int m = ot*16 + r + 8*grp;
            out[((long)b*Cout + m)*cNK + p0 + row] = acc[r];
        }
    }
}

// ============================ helpers ================================
__global__ void norms_kernel(const float* __restrict__ F, float* __restrict__ xx,
                             int C, long total)
{
    long t = blockIdx.x*256L + threadIdx.x;
    if (t >= total) return;
    int n = (int)(t % cN), b = (int)(t / cN);
    float s = 0.f;
    for (int c = 0; c < C; ++c) { float v = F[((long)b*C + c)*cN + n]; s += v*v; }
    xx[t] = s;
}

__global__ void topk_kernel(const float* __restrict__ S, const float* __restrict__ xx,
                            int* __restrict__ idx, long total)
{
    long t = blockIdx.x*256L + threadIdx.x;
    if (t >= total) return;
    int n = (int)(t % cN), b = (int)(t / cN);
    const float* Sr = S + ((long)b*cN + n)*cN;
    const float* xb = xx + (long)b*cN;
    float xn = xb[n];
    float bv[cK]; int bi[cK];
    for (int i = 0; i < cK; ++i) { bv[i] = -3.0e38f; bi[i] = 0; }
    for (int m = 0; m < cN; ++m) {
        float d = 2.f*Sr[m] - xn - xb[m];
        if (d > bv[cK-1]) {
            int p = cK-1;
            while (p > 0 && bv[p-1] < d) { bv[p] = bv[p-1]; bi[p] = bi[p-1]; --p; }
            bv[p] = d; bi[p] = m;
        }
    }
    int* op = idx + t*cK;
    for (int i = 0; i < cK; ++i) op[i] = bi[i];
}

__global__ void efr_fm(const float* __restrict__ F, const int* __restrict__ idx,
                       float* __restrict__ fm, int C, long total)
{
    long t = blockIdx.x*256L + threadIdx.x;
    if (t >= total) return;
    int b = (int)(t / cNK);
    int j = idx[t];
    float s = 0.f;
    for (int c = 0; c < C; ++c) s += F[((long)b*C + c)*cN + j];
    fm[t] = s / (float)C;
}

__global__ void channel_stats(const float* __restrict__ X, float* __restrict__ mean,
                              float* __restrict__ rstd, int C, size_t M, int Bc)
{
    __shared__ float ss[256], sq[256];
    int c = blockIdx.x, tid = threadIdx.x;
    float s = 0.f, q = 0.f;
    for (int b = 0; b < Bc; ++b) {
        const float* p = X + ((size_t)b*C + c)*M;
        for (size_t m = tid; m < M; m += 256) { float v = p[m]; s += v; q += v*v; }
    }
    ss[tid] = s; sq[tid] = q; __syncthreads();
    for (int o = 128; o > 0; o >>= 1) {
        if (tid < o) { ss[tid] += ss[tid+o]; sq[tid] += sq[tid+o]; }
        __syncthreads();
    }
    if (tid == 0) {
        float cnt = (float)((double)M * Bc);
        float mu  = ss[0] / cnt;
        float var = sq[0] / cnt - mu*mu;
        mean[c] = mu; rstd[c] = rsqrtf(fmaxf(var, 0.f) + 1e-5f);
    }
}

__global__ void bn_apply(float* __restrict__ X, const float* __restrict__ mean,
                         const float* __restrict__ rstd, int C, size_t M,
                         long total, int do_l)
{
    long t = blockIdx.x*256L + threadIdx.x;
    if (t >= total) return;
    int c = (int)((t / (long)M) % C);
    float v = (X[t] - mean[c]) * rstd[c];
    if (do_l) v = lrelu02(v);
    X[t] = v;
}

__global__ void attn_pre(const float* __restrict__ xq, const float* __restrict__ pos,
                         const float* __restrict__ Wp2, const float* __restrict__ Wef,
                         const float* __restrict__ fmf, const float* __restrict__ Wep,
                         const float* __restrict__ fmp, float* __restrict__ attn,
                         int fc, long total)
{
    long t = blockIdx.x*256L + threadIdx.x;
    if (t >= total) return;
    long nk = t % cNK; int b = (int)(t / cNK); int n = (int)(nk / cK);
    float fp = fmp[t], ff = fmf[t];
    float tmp[6];
    #pragma unroll
    for (int i = 0; i < 6; ++i) tmp[i] = pos[((long)b*6 + i)*cN + n] - Wep[i]*fp;
    float inv = rsqrtf((float)fc);
    for (int j = 0; j < fc; ++j) {
        float pj = 0.f;
        #pragma unroll
        for (int i = 0; i < 6; ++i) pj += Wp2[j*6 + i]*tmp[i];
        attn[((long)b*fc + j)*cNK + nk] = (xq[((long)b*fc + j)*cN + n] - Wef[j]*ff + pj) * inv;
    }
}

__global__ void softmax_k(float* __restrict__ a, long total)
{
    long t = blockIdx.x*256L + threadIdx.x;
    if (t >= total) return;
    float* p = a + t*cK;
    float mx = p[0];
    for (int i = 1; i < cK; ++i) mx = fmaxf(mx, p[i]);
    float s = 0.f;
    for (int i = 0; i < cK; ++i) { float e = expf(p[i]-mx); p[i] = e; s += e; }
    float inv = 1.f/s;
    for (int i = 0; i < cK; ++i) p[i] *= inv;
}

__global__ void wr_stats(const float* __restrict__ Wr, const float* __restrict__ Wf,
                         const float* __restrict__ fm_mean, const float* __restrict__ fm_rstd,
                         float* __restrict__ wre, float* __restrict__ m3,
                         float* __restrict__ r3, int fc)
{
    int o = threadIdx.x;
    if (o >= 64) return;
    float w = 0.f;
    for (int c = 0; c < fc; ++c) w += Wr[o*fc + c]*Wf[c];
    float mu  = fm_mean[0];
    float rs  = fm_rstd[0];
    float var = 1.f/(rs*rs) - 1e-5f;
    wre[o] = w; m3[o] = w*mu; r3[o] = rsqrtf(w*w*var + 1e-5f);
}

__global__ void gct_final(const float* __restrict__ y2, const float* __restrict__ m2,
                          const float* __restrict__ r2, const float* __restrict__ fmf,
                          const float* __restrict__ wre, const float* __restrict__ m3,
                          const float* __restrict__ r3, float* __restrict__ xout, long total)
{
    long t = blockIdx.x*256L + threadIdx.x;
    if (t >= total) return;
    long bo = t / cN; int n = (int)(t % cN);
    int o = (int)(bo % 64), b = (int)(bo / 64);
    const float* yp = y2 + bo*cNK + (long)n*cK;
    const float* fp = fmf + (long)b*cNK + (long)n*cK;
    float w = wre[o], mm2 = m2[o], rr2 = r2[o], mm3 = m3[o], rr3 = r3[o];
    float mx = -3.0e38f;
    for (int kk = 0; kk < cK; ++kk) {
        float v = (yp[kk]-mm2)*rr2 + (w*fp[kk]-mm3)*rr3;
        mx = fmaxf(mx, lrelu02(v));
    }
    xout[t] = mx;
}

__global__ void bn_lrelu_max(const float* __restrict__ y, const float* __restrict__ mean,
                             const float* __restrict__ rstd, float* __restrict__ xout,
                             int C, long total)
{
    long t = blockIdx.x*256L + threadIdx.x;
    if (t >= total) return;
    int o = (int)((t / cN) % C);
    const float* p = y + (t / cN)*cNK + (long)(t % cN)*cK;
    float m = mean[o], r = rstd[o];
    float mx = -3.0e38f;
    for (int kk = 0; kk < cK; ++kk) mx = fmaxf(mx, lrelu02((p[kk]-m)*r));
    xout[t] = mx;
}

__global__ void concat_cat(const float* __restrict__ x1, const float* __restrict__ x2,
                           const float* __restrict__ x3, const float* __restrict__ x4,
                           float* __restrict__ cat, long total)
{
    long t = blockIdx.x*256L + threadIdx.x;
    if (t >= total) return;
    int n = (int)(t % cN); long bc = t / cN;
    int c = (int)(bc % 512), b = (int)(bc / 512);
    float v;
    if (c < 64)       v = x1[((long)b*64  + c      )*cN + n];
    else if (c < 128) v = x2[((long)b*64  + (c-64) )*cN + n];
    else if (c < 256) v = x3[((long)b*128 + (c-128))*cN + n];
    else              v = x4[((long)b*256 + (c-256))*cN + n];
    cat[t] = v;
}

__global__ void pool_maxmean(const float* __restrict__ c5, float* __restrict__ z, long total)
{
    long t = blockIdx.x*256L + threadIdx.x;
    if (t >= total) return;
    int j = (int)(t % 2048), b = (int)(t / 2048);
    if (j < 1024) {
        const float* p = c5 + ((long)b*1024 + j)*cN;
        float mx = p[0];
        for (int n = 1; n < cN; ++n) mx = fmaxf(mx, p[n]);
        z[t] = mx;
    } else {
        const float* p = c5 + ((long)b*1024 + (j-1024))*cN;
        float s = 0.f;
        for (int n = 0; n < cN; ++n) s += p[n];
        z[t] = s / (float)cN;
    }
}

__global__ void fc_layer(const float* __restrict__ zin, const float* __restrict__ W,
                         const float* __restrict__ bias, float* __restrict__ zout,
                         int I, int O, long total)
{
    long t = blockIdx.x*256L + threadIdx.x;
    if (t >= total) return;
    int o = (int)(t % O), b = (int)(t / O);
    float s = bias ? bias[o] : 0.f;
    const float* zp = zin + (long)b*I;
    const float* wp = W + (long)o*I;
    for (int i = 0; i < I; ++i) s += zp[i]*wp[i];
    zout[t] = s;
}

// ======================= workspace layout ===========================
constexpr size_t alignup(size_t x) { return (x + 255) & ~(size_t)255; }
constexpr size_t SZ_IDX   = alignup((size_t)cB*cN*cK*4);
constexpr size_t OFF_IDX1 = 0;
constexpr size_t OFF_IDX2 = OFF_IDX1 + SZ_IDX;
constexpr size_t OFF_IDX3 = OFF_IDX2 + SZ_IDX;
constexpr size_t OFF_IDX4 = OFF_IDX3 + SZ_IDX;
constexpr size_t OFF_XX   = OFF_IDX4 + SZ_IDX;
constexpr size_t OFF_FM1  = OFF_XX   + alignup((size_t)cB*cN*4);
constexpr size_t OFF_FM2  = OFF_FM1  + SZ_IDX;
constexpr size_t OFF_FMP2 = OFF_FM2  + SZ_IDX;
constexpr size_t OFF_XQ   = OFF_FMP2 + SZ_IDX;
constexpr size_t OFF_POS  = OFF_XQ   + alignup((size_t)cB*128*cN*4);
constexpr size_t OFF_X1   = OFF_POS  + alignup((size_t)cB*6*cN*4);
constexpr size_t OFF_X2   = OFF_X1   + alignup((size_t)cB*64*cN*4);
constexpr size_t OFF_X3   = OFF_X2   + alignup((size_t)cB*64*cN*4);
constexpr size_t OFF_X4   = OFF_X3   + alignup((size_t)cB*128*cN*4);
constexpr size_t OFF_Z0   = OFF_X4   + alignup((size_t)cB*256*cN*4);
constexpr size_t OFF_Z1   = OFF_Z0   + alignup((size_t)cB*2048*4);
constexpr size_t OFF_Z2   = OFF_Z1   + alignup((size_t)cB*512*4);
constexpr size_t OFF_STA_M = OFF_Z2    + alignup((size_t)cB*256*4);
constexpr size_t OFF_STA_R = OFF_STA_M + 4096;
constexpr size_t OFF_STB_M = OFF_STA_R + 4096;
constexpr size_t OFF_STB_R = OFF_STB_M + 4096;
constexpr size_t OFF_STC_M = OFF_STB_R + 4096;
constexpr size_t OFF_STC_R = OFF_STC_M + 4096;
constexpr size_t OFF_WREF  = OFF_STC_R + 4096;
constexpr size_t OFF_AR0   = OFF_WREF  + 4096;                   // 84 MiB: S / attn / convout / cat
constexpr size_t OFF_AR1   = OFF_AR0 + (size_t)84*1024*1024;     // 42 MiB: xo / c5
constexpr size_t OFF_AR2   = OFF_AR1 + (size_t)42*1024*1024;     // 42 MiB: y2 (+Wc4 spill)

static inline int cdiv(long a, long b) { return (int)((a + b - 1) / b); }

extern "C" void kernel_launch(void* const* d_in, const int* in_sizes, int n_in,
                              void* d_out, int out_size, void* d_ws, size_t ws_size,
                              hipStream_t stream)
{
    (void)in_sizes; (void)n_in; (void)out_size; (void)ws_size;
    const float* x     = (const float*)d_in[0];
    const float* Wefr1 = (const float*)d_in[1];
    const float* Wefr2 = (const float*)d_in[2];
    const float* Wq1   = (const float*)d_in[3];
    const float* Wp1a  = (const float*)d_in[4];
    const float* Wp2a  = (const float*)d_in[5];
    const float* Wa1   = (const float*)d_in[6];
    const float* Wl1g  = (const float*)d_in[7];
    const float* Wr1   = (const float*)d_in[8];
    const float* Wq2   = (const float*)d_in[9];
    const float* Wp1b  = (const float*)d_in[10];
    const float* Wp2b  = (const float*)d_in[11];
    const float* Wa2   = (const float*)d_in[12];
    const float* Wl2g  = (const float*)d_in[13];
    const float* Wr2   = (const float*)d_in[14];
    const float* Wc3   = (const float*)d_in[15];
    const float* Wc4   = (const float*)d_in[16];
    const float* Wc5   = (const float*)d_in[17];
    const float* Wfc1  = (const float*)d_in[18];
    const float* Wfc2  = (const float*)d_in[19];
    const float* bfc2  = (const float*)d_in[20];
    const float* Wfc3  = (const float*)d_in[21];
    const float* bfc3  = (const float*)d_in[22];
    float* out = (float*)d_out;

    char* ws = (char*)d_ws;
    int*   idx1 = (int*)(ws + OFF_IDX1);
    int*   idx2 = (int*)(ws + OFF_IDX2);
    int*   idx3 = (int*)(ws + OFF_IDX3);
    int*   idx4 = (int*)(ws + OFF_IDX4);
    float* xxp  = (float*)(ws + OFF_XX);
    float* fm1  = (float*)(ws + OFF_FM1);
    float* fm2  = (float*)(ws + OFF_FM2);
    float* fmp2 = (float*)(ws + OFF_FMP2);
    float* xq   = (float*)(ws + OFF_XQ);
    float* pos  = (float*)(ws + OFF_POS);
    float* x1   = (float*)(ws + OFF_X1);
    float* x2   = (float*)(ws + OFF_X2);
    float* x3   = (float*)(ws + OFF_X3);
    float* x4   = (float*)(ws + OFF_X4);
    float* z0   = (float*)(ws + OFF_Z0);
    float* z1   = (float*)(ws + OFF_Z1);
    float* z2   = (float*)(ws + OFF_Z2);
    float* stA_m = (float*)(ws + OFF_STA_M);
    float* stA_r = (float*)(ws + OFF_STA_R);
    float* stB_m = (float*)(ws + OFF_STB_M);
    float* stB_r = (float*)(ws + OFF_STB_R);
    float* stC_m = (float*)(ws + OFF_STC_M);
    float* stC_r = (float*)(ws + OFF_STC_R);
    float* wre   = (float*)(ws + OFF_WREF);
    float* AR0   = (float*)(ws + OFF_AR0);   // S / attn / conv out / cat
    float* AR1   = (float*)(ws + OFF_AR1);   // xo / c5
    float* AR2   = (float*)(ws + OFF_AR2);   // y2

    // ---- kNN: norms -> WMMA inner-product GEMM -> top-k ----
    auto knn = [&](const float* F, int C, int* idxout) {
        norms_kernel<<<cdiv((long)cB*cN, 256), 256, 0, stream>>>(F, xxp, C, (long)cB*cN);
        dim3 g(cN/16, cN/16, cB);
        wmma_gemm_strided<<<g, 32, 0, stream>>>(F, F, AR0, cN, C, cN,
            (long)C*cN, 1L, (long)cN, (long)C*cN, (long)cN, (long)cN*cN, (long)cN);
        topk_kernel<<<cdiv((long)cB*cN, 256), 256, 0, stream>>>(AR0, xxp, idxout, (long)cB*cN);
    };

    // ---- graph-conv-transformer block ----
    auto gct = [&](const float* feat, int inC, const float* Wq, const float* Wp1,
                   const float* Wp2, const float* Wa, const float* Wl, const float* Wr,
                   const float* Wef, const float* fmf, const float* fmp, int fc,
                   float* xout) {
        // xq = lrelu(bn(c1d(Wq, feat)))
        { dim3 g(cN/16, cdiv(fc, 16), cB);
          wmma_gemm_strided<<<g, 32, 0, stream>>>(Wq, feat, xq, fc, inC, cN,
              0L, (long)inC, 1L, (long)inC*cN, (long)cN, (long)fc*cN, (long)cN); }
        channel_stats<<<fc, 256, 0, stream>>>(xq, stA_m, stA_r, fc, (size_t)cN, cB);
        bn_apply<<<cdiv((long)cB*fc*cN, 256), 256, 0, stream>>>(xq, stA_m, stA_r, fc,
            (size_t)cN, (long)cB*fc*cN, 1);
        // pos = lrelu(bn(c1d(Wp1, points)))
        { dim3 g(cN/16, 1, cB);
          wmma_gemm_strided<<<g, 32, 0, stream>>>(Wp1, x, pos, 6, 3, cN,
              0L, 3L, 1L, (long)3*cN, (long)cN, (long)6*cN, (long)cN); }
        channel_stats<<<6, 256, 0, stream>>>(pos, stB_m, stB_r, 6, (size_t)cN, cB);
        bn_apply<<<cdiv((long)cB*6*cN, 256), 256, 0, stream>>>(pos, stB_m, stB_r, 6,
            (size_t)cN, (long)cB*6*cN, 1);
        // attention scores + softmax over k
        attn_pre<<<cdiv((long)cB*cNK, 256), 256, 0, stream>>>(xq, pos, Wp2, Wef, fmf,
            Wefr1, fmp, AR0, fc, (long)cB*cNK);
        softmax_k<<<cdiv((long)cB*fc*cN, 256), 256, 0, stream>>>(AR0, (long)cB*fc*cN);
        // fused Wa GEMM + value contraction -> xo (B,64,N,K)
        { dim3 g((unsigned)(cNK/16), cB);
          gct_xo_wmma<<<g, 32, 0, stream>>>(Wa, AR0, Wefr1, fmp, AR1, fc); }
        channel_stats<<<64, 256, 0, stream>>>(AR1, stA_m, stA_r, 64, (size_t)cNK, cB);
        bn_apply<<<cdiv((long)cB*64*cNK, 256), 256, 0, stream>>>(AR1, stA_m, stA_r, 64,
            (size_t)cNK, (long)cB*64*cNK, 1);
        // y2 = Wl @ y
        { dim3 g((unsigned)(cNK/16), 4, cB);
          wmma_gemm_strided<<<g, 32, 0, stream>>>(Wl, AR1, AR2, 64, 64, (int)cNK,
              0L, 64L, 1L, (long)64*cNK, cNK, (long)64*cNK, cNK); }
        channel_stats<<<64, 256, 0, stream>>>(AR2, stA_m, stA_r, 64, (size_t)cNK, cB);
        // y3 branch is rank-1: Wr@knn_f = (Wr@Wef) * fm -> analytic bn stats
        channel_stats<<<1, 256, 0, stream>>>(fmf, stB_m, stB_r, 1, (size_t)cNK, cB);
        wr_stats<<<1, 64, 0, stream>>>(Wr, Wef, stB_m, stB_r, wre, stC_m, stC_r, fc);
        gct_final<<<cdiv((long)cB*64*cN, 256), 256, 0, stream>>>(AR2, stA_m, stA_r,
            fmf, wre, stC_m, stC_r, xout, (long)cB*64*cN);
    };

    // ---- edge-conv (ggf) block ----
    auto ggfconv = [&](const float* F, int Cf, const float* Wc, int Cout,
                       const int* idxp, float* xout) {
        { dim3 g((unsigned)(cNK/16), cB);
          ggf_conv_wmma<<<g, 32, 0, stream>>>(Wc, F, idxp, AR0, Cf, Cout); }
        channel_stats<<<Cout, 256, 0, stream>>>(AR0, stA_m, stA_r, Cout, (size_t)cNK, cB);
        bn_lrelu_max<<<cdiv((long)cB*Cout*cN, 256), 256, 0, stream>>>(AR0, stA_m, stA_r,
            xout, Cout, (long)cB*Cout*cN);
    };

    // ================= pipeline =================
    knn(x, 3, idx1);
    efr_fm<<<cdiv((long)cB*cNK, 256), 256, 0, stream>>>(x, idx1, fm1, 3, (long)cB*cNK);
    gct(x, 3, Wq1, Wp1a, Wp2a, Wa1, Wl1g, Wr1, Wefr1, fm1, fm1, 6, x1);

    knn(x1, 64, idx2);
    efr_fm<<<cdiv((long)cB*cNK, 256), 256, 0, stream>>>(x1, idx2, fm2, 64, (long)cB*cNK);
    efr_fm<<<cdiv((long)cB*cNK, 256), 256, 0, stream>>>(x,  idx2, fmp2, 3, (long)cB*cNK);
    gct(x1, 64, Wq2, Wp1b, Wp2b, Wa2, Wl2g, Wr2, Wefr2, fm2, fmp2, 128, x2);

    knn(x2, 64, idx3);
    ggfconv(x2, 64, Wc3, 128, idx3, x3);

    knn(x3, 128, idx4);
    ggfconv(x3, 128, Wc4, 256, idx4, x4);

    // cat -> Wc5 -> bn/lrelu -> pools
    float* cat = AR0;
    float* c5  = AR1;
    concat_cat<<<cdiv((long)cB*512*cN, 256), 256, 0, stream>>>(x1, x2, x3, x4, cat,
        (long)cB*512*cN);
    { dim3 g(cN/16, 64, cB);
      wmma_gemm_strided<<<g, 32, 0, stream>>>(Wc5, cat, c5, 1024, 512, cN,
          0L, 512L, 1L, (long)512*cN, (long)cN, (long)1024*cN, (long)cN); }
    channel_stats<<<1024, 256, 0, stream>>>(c5, stA_m, stA_r, 1024, (size_t)cN, cB);
    bn_apply<<<cdiv((long)cB*1024*cN, 256), 256, 0, stream>>>(c5, stA_m, stA_r, 1024,
        (size_t)cN, (long)cB*1024*cN, 1);
    pool_maxmean<<<cdiv((long)cB*2048, 256), 256, 0, stream>>>(c5, z0, (long)cB*2048);

    // FC head
    fc_layer<<<cdiv((long)cB*512, 256), 256, 0, stream>>>(z0, Wfc1, (const float*)nullptr,
        z1, 2048, 512, (long)cB*512);
    channel_stats<<<512, 256, 0, stream>>>(z1, stA_m, stA_r, 512, (size_t)1, cB);
    bn_apply<<<cdiv((long)cB*512, 256), 256, 0, stream>>>(z1, stA_m, stA_r, 512,
        (size_t)1, (long)cB*512, 1);
    fc_layer<<<cdiv((long)cB*256, 256), 256, 0, stream>>>(z1, Wfc2, bfc2, z2, 512, 256,
        (long)cB*256);
    channel_stats<<<256, 256, 0, stream>>>(z2, stA_m, stA_r, 256, (size_t)1, cB);
    bn_apply<<<cdiv((long)cB*256, 256), 256, 0, stream>>>(z2, stA_m, stA_r, 256,
        (size_t)1, (long)cB*256, 1);
    fc_layer<<<cdiv((long)cB*40, 256), 256, 0, stream>>>(z2, Wfc3, bfc3, out, 256, 40,
        (long)cB*40);
}
